// UVMB_7267084664863
// MI455X (gfx1250) — matrix operand order repaired and checked by reading, not compiled, not run on hardware
//
#include <hip/hip_runtime.h>
#include <hip/hip_bf16.h>
#include <math.h>

// ---------------- problem constants ----------------
#define BATCH   2
#define CIN     3
#define TT      8
#define WW      96
#define HH      96
#define NPOS    (TT*WW*HH)        // 73728 spatial positions per batch
#define C1OUT   16
#define LSEQ    NPOS              // Mamba sequence length per batch
#define DINNER  6
#define DSTATE  16
#define DCONV   4
#define NCH     (DINNER*DSTATE)   // 96 scan channels per batch
#define CHUNK   1024              // scan chunk length
#define NCHUNK  (LSEQ/CHUNK)      // 72 chunks per batch
#define CBLK    192               // conv block: 6 waves = 6 h-tiles of one (b,t,w)

typedef __attribute__((ext_vector_type(16))) _Float16 v16h;
typedef __attribute__((ext_vector_type(8)))  float    v8f;

__device__ __forceinline__ float siluf(float x) { return x / (1.f + __expf(-x)); }

// In-step A-matrix tap index (16x32 f16 A operand), element e, given half:
//   tau(e, half) = base(e) + 8*half,  base = v<4 ? 2v+p : 2v+p+8   (v=e>>1,p=e&1)
// We enumerate K as k' = ci*32 + tau (tau<27 valid), so each WMMA step is one ci.

// ============================================================
// Kernel 1: conv3d 3->16 (k=3^3, same) + bias + ReLU via WMMA.
// Block = one (b,t,w) row, 6 waves = 6 h-tiles.  K' = 3*32.
// Input patch and f16 B image staged in LDS.
// ============================================================
__global__ void k_conv1_wmma(const float* __restrict__ x,
                             const float* __restrict__ Wc,   // (16,3,3,3,3)
                             const float* __restrict__ bc,   // (16)
                             float* __restrict__ out)        // (2,16,NPOS)
{
  __shared__ __align__(32) _Float16 wlds[16 * 96];  // [n][k'=ci*32+tau]
  __shared__ float lin[27 * 98];                    // [ci][td][wd][hh(98)] zero-padded
  int tid = threadIdx.x;

  int w = blockIdx.x % WW;
  int t = (blockIdx.x / WW) % TT;
  int b =  blockIdx.x / (WW * TT);
  const float* xb = x + (size_t)b * CIN * NPOS;

  // stage weights (k' order, pad tau 27..31 with 0)
  for (int i = tid; i < 16 * 96; i += CBLK) {
    int n = i / 96, kp = i % 96, ci = kp >> 5, tau = kp & 31;
    wlds[i] = (_Float16)((tau < 27) ? Wc[n * 81 + ci * 27 + tau] : 0.f);
  }
  // stage zero-padded input patch: 27 rows of 98
  for (int i = tid; i < 27 * 98; i += CBLK) {
    int row = i / 98, hh = i % 98;
    int wd = row % 3, tmp = row / 3, td = tmp % 3, ci = tmp / 3;
    int tt = t + td - 1, wq = w + wd - 1, hq = hh - 1;
    bool ok = tt >= 0 && tt < TT && wq >= 0 && wq < WW && hq >= 0 && hq < HH;
    float v = xb[((ci * TT + min(max(tt, 0), TT - 1)) * WW +
                  min(max(wq, 0), WW - 1)) * HH + min(max(hq, 0), HH - 1)];
    lin[i] = ok ? v : 0.f;
  }
  __syncthreads();

  int wid  = tid >> 5;          // 0..5 -> h-tile
  int lane = tid & 31;
  int nh   = lane & 15;         // A row m / B,C column n
  int half = lane >> 4;
  int h0   = wid * 16;
  int hb   = h0 + nh;

  v8f c = {};
  #pragma unroll
  for (int step = 0; step < 3; ++step) {          // step == ci
    v16h bf;
    for (int s = 0; s < 16; ++s) bf[s] = wlds[nh * 96 + step * 32 + 8 * half + s + (s & 8)];
    const float* base = &lin[step * 9 * 98];
    v16h af;
    #pragma unroll
    for (int e = 0; e < 16; ++e) {
      const int v = e >> 1, p = e & 1;
      const int t0_ = (v < 4) ? (2 * v + p) : (2 * v + p + 8);  // tau, half 0
      const int t1_ = t0_ + 8;                                  // tau, half 1
      const int t1c = (t1_ > 26) ? 26 : t1_;
      const int offA = ((t0_ / 9) * 3 + ((t0_ % 9) / 3)) * 98 + (t0_ % 3);
      const int offB = ((t1c / 9) * 3 + ((t1c % 9) / 3)) * 98 + (t1_ % 3);
      int off = (half ? offB : offA) + hb;        // hh index = hq+1 (pad in LDS)
      float val = base[off];
      float res = (t1_ >= 27 && half) ? 0.f : val;  // only pad taps need a mask
      af[e] = (_Float16)res;
    }
    c = __builtin_amdgcn_wmma_f32_16x16x32_f16(false, af, false, bf,
                                               (short)0, c, false, false);
  }
  float bias = bc[nh];
  float4 s0, s1;
  s0.x = fmaxf(c[0] + bias, 0.f); s0.y = fmaxf(c[1] + bias, 0.f);
  s0.z = fmaxf(c[2] + bias, 0.f); s0.w = fmaxf(c[3] + bias, 0.f);
  s1.x = fmaxf(c[4] + bias, 0.f); s1.y = fmaxf(c[5] + bias, 0.f);
  s1.z = fmaxf(c[6] + bias, 0.f); s1.w = fmaxf(c[7] + bias, 0.f);
  size_t ob = (size_t)(b * C1OUT + nh) * NPOS + (t * WW + w) * HH + h0 + 8 * half;
  *(float4*)&out[ob]     = s0;
  *(float4*)&out[ob + 4] = s1;
}

// ============================================================
// Kernel 2: conv3d 16->3 + bias + residual(x) via WMMA.
// K' = 16*32 = 512, 16 steps (one ci each); input staged in LDS
// 4 channels per round (4 rounds, barrier-recycled buffer).
// ============================================================
__global__ void k_conv2_wmma(const float* __restrict__ mid,  // (2,16,NPOS)
                             const float* __restrict__ Wc,   // (3,16,3,3,3)
                             const float* __restrict__ bc,   // (3)
                             const float* __restrict__ x,    // (2,3,NPOS)
                             float* __restrict__ xr)         // (2,3,NPOS)
{
  __shared__ __align__(32) _Float16 wlds[16 * 512];  // [n][k'=ci*32+tau]
  __shared__ float lin[36 * 98];                     // 4 ci x [td][wd][hh]
  int tid = threadIdx.x;

  int w = blockIdx.x % WW;
  int t = (blockIdx.x / WW) % TT;
  int b =  blockIdx.x / (WW * TT);
  const float* mb = mid + (size_t)b * C1OUT * NPOS;

  for (int i = tid; i < 16 * 512; i += CBLK) {
    int n = i >> 9, kp = i & 511, ci = kp >> 5, tau = kp & 31;
    wlds[i] = (_Float16)((tau < 27 && n < 3) ? Wc[n * 432 + ci * 27 + tau] : 0.f);
  }

  int wid  = tid >> 5;
  int lane = tid & 31;
  int nh   = lane & 15;
  int half = lane >> 4;
  int h0   = wid * 16;
  int hb   = h0 + nh;

  v8f c = {};
  for (int r = 0; r < 4; ++r) {
    __syncthreads();                               // LDS buffer reuse
    for (int i = tid; i < 36 * 98; i += CBLK) {
      int row = i / 98, hh = i % 98;
      int wd = row % 3, tmp = row / 3, td = tmp % 3, cil = tmp / 3;
      int ci = r * 4 + cil;
      int tt = t + td - 1, wq = w + wd - 1, hq = hh - 1;
      bool ok = tt >= 0 && tt < TT && wq >= 0 && wq < WW && hq >= 0 && hq < HH;
      float v = mb[((ci * TT + min(max(tt, 0), TT - 1)) * WW +
                    min(max(wq, 0), WW - 1)) * HH + min(max(hq, 0), HH - 1)];
      lin[i] = ok ? v : 0.f;
    }
    __syncthreads();
    #pragma unroll
    for (int sl = 0; sl < 4; ++sl) {
      int step = r * 4 + sl;                       // global k'-step (== ci)
      v16h bf;
      for (int s = 0; s < 16; ++s) bf[s] = wlds[nh * 512 + step * 32 + 8 * half + s + (s & 8)];
      const float* base = &lin[sl * 9 * 98];
      v16h af;
      #pragma unroll
      for (int e = 0; e < 16; ++e) {
        const int v = e >> 1, p = e & 1;
        const int t0_ = (v < 4) ? (2 * v + p) : (2 * v + p + 8);
        const int t1_ = t0_ + 8;
        const int t1c = (t1_ > 26) ? 26 : t1_;
        const int offA = ((t0_ / 9) * 3 + ((t0_ % 9) / 3)) * 98 + (t0_ % 3);
        const int offB = ((t1c / 9) * 3 + ((t1c % 9) / 3)) * 98 + (t1_ % 3);
        int off = (half ? offB : offA) + hb;
        float val = base[off];
        float res = (t1_ >= 27 && half) ? 0.f : val;
        af[e] = (_Float16)res;
      }
      c = __builtin_amdgcn_wmma_f32_16x16x32_f16(false, af, false, bf,
                                                 (short)0, c, false, false);
    }
  }
  if (nh < 3) {
    float bias = bc[nh];
    size_t ob = (size_t)(b * CIN + nh) * NPOS + (t * WW + w) * HH + h0 + 8 * half;
    float4 r0 = *(const float4*)&x[ob];
    float4 r1 = *(const float4*)&x[ob + 4];
    float4 s0, s1;
    s0.x = c[0] + bias + r0.x; s0.y = c[1] + bias + r0.y;
    s0.z = c[2] + bias + r0.z; s0.w = c[3] + bias + r0.w;
    s1.x = c[4] + bias + r1.x; s1.y = c[5] + bias + r1.y;
    s1.z = c[6] + bias + r1.z; s1.w = c[7] + bias + r1.w;
    *(float4*)&xr[ob]     = s0;
    *(float4*)&xr[ob + 4] = s1;
  }
}

// ============================================================
// Kernel 3: per-token Mamba front-end.
// token l of batch b reads xr_flat[b][3l..3l+2]  (the faithful reshape).
// Produces channel-major SoA: xc,dt,silu(z) (b,6,L); B,C (b,16,L).
// ============================================================
__global__ void k_mamba_pre(const float* __restrict__ xr,
                            const float* __restrict__ W_in,    // (12,3)
                            const float* __restrict__ W_conv,  // (6,1,4)
                            const float* __restrict__ b_conv,  // (6)
                            const float* __restrict__ W_xproj, // (33,6)
                            const float* __restrict__ W_dt,    // (6,1)
                            const float* __restrict__ b_dt,    // (6)
                            float* __restrict__ xcA, float* __restrict__ dtA,
                            float* __restrict__ zgA, float* __restrict__ BA,
                            float* __restrict__ CA)
{
  int idx = blockIdx.x * blockDim.x + threadIdx.x;
  if (idx >= BATCH * LSEQ) return;
  int b = idx / LSEQ, l = idx % LSEQ;
  const float* xb = xr + (size_t)b * (CIN * NPOS);

  float xs[DCONV][DINNER];
  #pragma unroll
  for (int q = 0; q < DCONV; ++q) {
    int lp = l - (DCONV - 1) + q;
    float r0 = 0.f, r1 = 0.f, r2 = 0.f;
    if (lp >= 0) { r0 = xb[3 * lp]; r1 = xb[3 * lp + 1]; r2 = xb[3 * lp + 2]; }
    #pragma unroll
    for (int d = 0; d < DINNER; ++d)
      xs[q][d] = W_in[d * 3] * r0 + W_in[d * 3 + 1] * r1 + W_in[d * 3 + 2] * r2;
  }
  float z[DINNER];
  {
    float r0 = xb[3 * l], r1 = xb[3 * l + 1], r2 = xb[3 * l + 2];
    #pragma unroll
    for (int d = 0; d < DINNER; ++d)
      z[d] = W_in[(DINNER + d) * 3] * r0 + W_in[(DINNER + d) * 3 + 1] * r1 +
             W_in[(DINNER + d) * 3 + 2] * r2;
  }
  float xc[DINNER];
  #pragma unroll
  for (int d = 0; d < DINNER; ++d) {
    float a = b_conv[d];
    #pragma unroll
    for (int k = 0; k < DCONV; ++k) a += W_conv[d * DCONV + k] * xs[k][d];
    xc[d] = siluf(a);
  }
  float xdbl[33];
  #pragma unroll
  for (int i = 0; i < 33; ++i) {
    float a = 0.f;
    #pragma unroll
    for (int d = 0; d < DINNER; ++d) a += W_xproj[i * DINNER + d] * xc[d];
    xdbl[i] = a;
  }
  float dtin = xdbl[0];
  #pragma unroll
  for (int d = 0; d < DINNER; ++d) {
    float u  = dtin * W_dt[d] + b_dt[d];
    float sp = (u > 20.f) ? u : log1pf(__expf(u));   // softplus
    size_t o = (size_t)(b * DINNER + d) * LSEQ + l;
    dtA[o] = sp;
    xcA[o] = xc[d];
    zgA[o] = siluf(z[d]);
  }
  #pragma unroll
  for (int s = 0; s < DSTATE; ++s) {
    size_t o = (size_t)(b * DSTATE + s) * LSEQ + l;
    BA[o] = xdbl[1 + s];
    CA[o] = xdbl[17 + s];
  }
}

// ============================================================
// Kernel 4: chunk-local scan (phase 1).
// ============================================================
__global__ void k_scan_local(const float* __restrict__ dtA,
                             const float* __restrict__ xcA,
                             const float* __restrict__ BA,
                             const float* __restrict__ A_log,  // (6,16)
                             float* __restrict__ aTot, float* __restrict__ hTot)
{
  int tid = threadIdx.x;                   // 0..95
  int g = blockIdx.x % NCHUNK, b = blockIdx.x / NCHUNK;
  int d = tid >> 4, s = tid & 15;
  float Ac = -__expf(A_log[d * DSTATE + s]);
  const float* dtp = dtA + (size_t)(b * DINNER + d) * LSEQ + g * CHUNK;
  const float* xcp = xcA + (size_t)(b * DINNER + d) * LSEQ + g * CHUNK;
  const float* Bp  = BA  + (size_t)(b * DSTATE + s) * LSEQ + g * CHUNK;
  float Ap = 1.f, h = 0.f;
  for (int t0 = 0; t0 < CHUNK; t0 += 64) {
    __builtin_prefetch(dtp + t0 + 64, 0, 1);   // global_prefetch_b8
    __builtin_prefetch(xcp + t0 + 64, 0, 1);
    __builtin_prefetch(Bp  + t0 + 64, 0, 1);
    for (int t = t0; t < t0 + 64; ++t) {
      float dt = dtp[t], xc = xcp[t], Bs = Bp[t];
      float a = __expf(dt * Ac);
      h  = a * h + dt * Bs * xc;
      Ap = Ap * a;
    }
  }
  aTot[(size_t)blockIdx.x * NCH + tid] = Ap;
  hTot[(size_t)blockIdx.x * NCH + tid] = h;
}

// ============================================================
// Kernel 5: cross-chunk carry scan (phase 2).
// ============================================================
__global__ void k_scan_carry(const float* __restrict__ aTot,
                             const float* __restrict__ hTot,
                             float* __restrict__ carry)
{
  int tid = threadIdx.x;
  if (tid >= BATCH * NCH) return;
  int b = tid / NCH, k = tid % NCH;
  float c = 0.f;
  for (int g = 0; g < NCHUNK; ++g) {
    size_t i = (size_t)(b * NCHUNK + g) * NCH + k;
    carry[i] = c;
    c = aTot[i] * c + hTot[i];
  }
}

// ============================================================
// Kernel 6: finalize (phase 3).
// ============================================================
__global__ void k_scan_final(const float* __restrict__ dtA,
                             const float* __restrict__ xcA,
                             const float* __restrict__ BA,
                             const float* __restrict__ CA,
                             const float* __restrict__ zgA,
                             const float* __restrict__ A_log,
                             const float* __restrict__ Dv,     // (6)
                             const float* __restrict__ W_out,  // (3,6)
                             const float* __restrict__ carry,
                             float* __restrict__ ymid)         // (2,3,L)
{
  __shared__ float ybuf[DINNER];
  int tid = threadIdx.x;                   // 0..95
  int g = blockIdx.x % NCHUNK, b = blockIdx.x / NCHUNK;
  int d = tid >> 4, s = tid & 15;
  float Ac = -__expf(A_log[d * DSTATE + s]);
  float Dd = Dv[d];
  float wo[DINNER];
  if (tid < 3) {
    #pragma unroll
    for (int q = 0; q < DINNER; ++q) wo[q] = W_out[tid * DINNER + q];
  }
  const float* dtp = dtA + (size_t)(b * DINNER + d) * LSEQ + g * CHUNK;
  const float* xcp = xcA + (size_t)(b * DINNER + d) * LSEQ + g * CHUNK;
  const float* zgp = zgA + (size_t)(b * DINNER + d) * LSEQ + g * CHUNK;
  const float* Bp  = BA  + (size_t)(b * DSTATE + s) * LSEQ + g * CHUNK;
  const float* Cp  = CA  + (size_t)(b * DSTATE + s) * LSEQ + g * CHUNK;
  float h = carry[(size_t)blockIdx.x * NCH + tid];

  for (int t0 = 0; t0 < CHUNK; t0 += 64) {
    __builtin_prefetch(dtp + t0 + 64, 0, 1);
    __builtin_prefetch(xcp + t0 + 64, 0, 1);
    __builtin_prefetch(zgp + t0 + 64, 0, 1);
    __builtin_prefetch(Bp  + t0 + 64, 0, 1);
    __builtin_prefetch(Cp  + t0 + 64, 0, 1);
    for (int t = t0; t < t0 + 64; ++t) {
      float dt = dtp[t], xc = xcp[t], Bs = Bp[t], Cs = Cp[t];
      float a = __expf(dt * Ac);
      h = a * h + dt * Bs * xc;
      float v = h * Cs;
      v += __shfl_xor(v, 1, 32);
      v += __shfl_xor(v, 2, 32);
      v += __shfl_xor(v, 4, 32);
      v += __shfl_xor(v, 8, 32);           // sum over s within 16-lane half
      if (s == 0) ybuf[d] = (v + xc * Dd) * zgp[t];
      __syncthreads();
      if (tid < 3) {
        float o = 0.f;
        #pragma unroll
        for (int q = 0; q < DINNER; ++q) o += wo[q] * ybuf[q];
        ymid[(size_t)(b * CIN + tid) * LSEQ + g * CHUNK + t] = o;
      }
      __syncthreads();
    }
  }
}

// ============================================================
// Kernel 7: final conv3d 3->3 (k=3^3, same) — tiny, plain VALU.
// ============================================================
__global__ void k_conv3(const float* __restrict__ ymid,       // (2,3,NPOS)
                        const float* __restrict__ Wc,         // (3,3,3,3,3)
                        const float* __restrict__ bc,         // (3)
                        float* __restrict__ out)              // (2,3,NPOS)
{
  int idx = blockIdx.x * blockDim.x + threadIdx.x;
  if (idx >= BATCH * CIN * NPOS) return;
  int b   = idx / (CIN * NPOS);
  int rem = idx % (CIN * NPOS);
  int j   = rem / NPOS;
  int pos = rem % NPOS;
  int t = pos / (WW * HH), w = (pos / HH) % WW, h = pos % HH;

  float acc = bc[j];
  #pragma unroll
  for (int ci = 0; ci < 3; ++ci)
    #pragma unroll
    for (int kd = 0; kd < 3; ++kd) {
      int tt = t + kd - 1;
      bool okt = (tt >= 0 && tt < TT);
      int ttc = min(max(tt, 0), TT - 1);
      #pragma unroll
      for (int kh = 0; kh < 3; ++kh) {
        int wq = w + kh - 1;
        bool okw = okt && (wq >= 0 && wq < WW);
        int wqc = min(max(wq, 0), WW - 1);
        #pragma unroll
        for (int kw = 0; kw < 3; ++kw) {
          int hq = h + kw - 1;
          bool ok = okw && (hq >= 0 && hq < HH);
          int hqc = min(max(hq, 0), HH - 1);
          float v = ymid[(size_t)(b * CIN + ci) * NPOS + (ttc * WW + wqc) * HH + hqc];
          acc += Wc[(j * 3 + ci) * 27 + kd * 9 + kh * 3 + kw] * (ok ? v : 0.f);
        }
      }
    }
  out[idx] = acc;
}

// ============================================================
extern "C" void kernel_launch(void* const* d_in, const int* in_sizes, int n_in,
                              void* d_out, int out_size, void* d_ws, size_t ws_size,
                              hipStream_t stream) {
  const float* x       = (const float*)d_in[0];
  const float* W_cb1   = (const float*)d_in[1];
  const float* b_cb1   = (const float*)d_in[2];
  const float* W_cb2   = (const float*)d_in[3];
  const float* b_cb2   = (const float*)d_in[4];
  const float* W_in    = (const float*)d_in[5];
  const float* W_conv  = (const float*)d_in[6];
  const float* b_conv  = (const float*)d_in[7];
  const float* W_xproj = (const float*)d_in[8];
  const float* W_dt    = (const float*)d_in[9];
  const float* b_dt    = (const float*)d_in[10];
  const float* A_log   = (const float*)d_in[11];
  const float* Dv      = (const float*)d_in[12];
  const float* W_out   = (const float*)d_in[13];
  const float* W_sm    = (const float*)d_in[14];
  const float* b_sm    = (const float*)d_in[15];
  float* out = (float*)d_out;

  // workspace partition (floats)
  float* ws = (float*)d_ws;
  size_t o = 0;
  float* buf1 = ws + o; o += (size_t)BATCH * C1OUT * NPOS;
  float* xr   = ws + o; o += (size_t)BATCH * CIN * NPOS;
  float* xcA  = ws + o; o += (size_t)BATCH * DINNER * LSEQ;
  float* dtA  = ws + o; o += (size_t)BATCH * DINNER * LSEQ;
  float* zgA  = ws + o; o += (size_t)BATCH * DINNER * LSEQ;
  float* BA   = ws + o; o += (size_t)BATCH * DSTATE * LSEQ;
  float* CA   = ws + o; o += (size_t)BATCH * DSTATE * LSEQ;
  float* ymid = ws + o; o += (size_t)BATCH * CIN * LSEQ;
  float* aTot = ws + o; o += (size_t)BATCH * NCHUNK * NCH;
  float* hTot = ws + o; o += (size_t)BATCH * NCHUNK * NCH;
  float* cary = ws + o; o += (size_t)BATCH * NCHUNK * NCH;

  dim3 cgrd(BATCH * TT * WW);                 // 1536 blocks: one per (b,t,w)
  dim3 cblk(CBLK);                            // 6 waves = 6 h-tiles

  k_conv1_wmma<<<cgrd, cblk, 0, stream>>>(x, W_cb1, b_cb1, buf1);
  k_conv2_wmma<<<cgrd, cblk, 0, stream>>>(buf1, W_cb2, b_cb2, x, xr);

  int ntok = BATCH * LSEQ;
  k_mamba_pre<<<(ntok + 255) / 256, 256, 0, stream>>>(
      xr, W_in, W_conv, b_conv, W_xproj, W_dt, b_dt, xcA, dtA, zgA, BA, CA);

  k_scan_local<<<BATCH * NCHUNK, NCH, 0, stream>>>(dtA, xcA, BA, A_log, aTot, hTot);
  k_scan_carry<<<1, BATCH * NCH, 0, stream>>>(aTot, hTot, cary);
  k_scan_final<<<BATCH * NCHUNK, NCH, 0, stream>>>(
      dtA, xcA, BA, CA, zgA, A_log, Dv, W_out, cary, ymid);

  int nout = BATCH * CIN * NPOS;
  k_conv3<<<(nout + 255) / 256, 256, 0, stream>>>(ymid, W_sm, b_sm, out);
}
